// Attention_37984690766206
// MI455X (gfx1250) — matrix operand order repaired and checked by reading, not compiled, not run on hardware
//
#include <hip/hip_runtime.h>
#include <hip/hip_bf16.h>

// Attention forward for MI455X (gfx1250).
// - All GEMMs + flash attention use v_wmma_f32_16x16x32_bf16 (wave32).
// - GEMM tiles streamed with CDNA5 async global->LDS copies
//   (global_load_async_to_lds_b128 / s_wait_asynccnt), double-buffered.
// - Weights pre-converted to bf16 transposed [N,K] so every WMMA fragment
//   load (from LDS or global) is a contiguous 16/32-byte run per lane.
// - Online softmax with ds_swizzle xor-reductions inside 16-lane N-groups.

#define BB   2
#define SS   2048
#define DDIM 2048
#define HH   16
#define KVH  4
#define HDIM 128
#define REP  (HH / KVH)

typedef __attribute__((ext_vector_type(16))) __bf16 v16bf;
typedef __attribute__((ext_vector_type(8)))  float  v8f;
typedef unsigned short u16;

__device__ __forceinline__ u16 f2bf(float f) {
  unsigned u = __float_as_uint(f);
  u += 0x7FFFu + ((u >> 16) & 1u);   // round-to-nearest-even
  return (u16)(u >> 16);
}

__device__ __forceinline__ v8f wmma_bf16(v16bf a, v16bf b, v8f c) {
  return __builtin_amdgcn_wmma_f32_16x16x32_bf16(false, a, false, b, (short)0, c,
                                                 false, false);
}

// ---- CDNA5 async global->LDS copy (ASYNCcnt), inline asm for portability ----
__device__ __forceinline__ void async_ld16(const u16* g, u16* lds) {
  unsigned l = (unsigned)(unsigned long long)lds;  // low 32 bits = LDS offset
  asm volatile("global_load_async_to_lds_b128 %0, %1, off"
               :: "v"(l), "v"(g) : "memory");
}
__device__ __forceinline__ void wait_async0() {
  asm volatile("s_wait_asynccnt 0" ::: "memory");
}

// xor-reductions within each 16-lane half of a wave32 (ds_swizzle group-of-32).
__device__ __forceinline__ float red_max16(float x) {
  x = fmaxf(x, __int_as_float(__builtin_amdgcn_ds_swizzle(__float_as_int(x), 0x041F)));
  x = fmaxf(x, __int_as_float(__builtin_amdgcn_ds_swizzle(__float_as_int(x), 0x081F)));
  x = fmaxf(x, __int_as_float(__builtin_amdgcn_ds_swizzle(__float_as_int(x), 0x101F)));
  x = fmaxf(x, __int_as_float(__builtin_amdgcn_ds_swizzle(__float_as_int(x), 0x201F)));
  return x;
}
__device__ __forceinline__ float red_sum16(float x) {
  x += __int_as_float(__builtin_amdgcn_ds_swizzle(__float_as_int(x), 0x041F));
  x += __int_as_float(__builtin_amdgcn_ds_swizzle(__float_as_int(x), 0x081F));
  x += __int_as_float(__builtin_amdgcn_ds_swizzle(__float_as_int(x), 0x101F));
  x += __int_as_float(__builtin_amdgcn_ds_swizzle(__float_as_int(x), 0x201F));
  return x;
}

// ---------------------------------------------------------------------------
// fp32 -> bf16 copy, and fp32 [K,N] -> bf16 transposed [N,K]
// ---------------------------------------------------------------------------
__global__ void to_bf16(const float* __restrict__ src, u16* __restrict__ dst, long n) {
  long i = (long)blockIdx.x * blockDim.x + threadIdx.x;
  if (i < n) dst[i] = f2bf(src[i]);
}
__global__ void w_transpose_bf16(const float* __restrict__ w, u16* __restrict__ wt,
                                 int K, int N) {
  long idx = (long)blockIdx.x * blockDim.x + threadIdx.x;
  if (idx >= (long)K * N) return;
  int  n = (int)(idx % N);
  long k = idx / N;
  wt[(long)n * K + k] = f2bf(w[idx]);
}

// ---------------------------------------------------------------------------
// GEMM: C[M,N] = A[M,K] @ Wt[N,K]^T ; A,Wt bf16, C bf16 or fp32.
// Block tile 128x64, 8 waves (4x2), wave -> 32x32 (2x2 WMMA accums).
// K-tiles of 32 streamed via async global->LDS copies, double-buffered.
// ---------------------------------------------------------------------------
template <bool OUT_BF16>
__global__ __launch_bounds__(256, 2)
void gemm_bf16(const u16* __restrict__ A, const u16* __restrict__ Wt,
               void* __restrict__ Cp, int M, int N, int K) {
  __shared__ u16 sA[2][128][40];  // [buf][m][k] (+pad, rows 80B = 16B-aligned)
  __shared__ u16 sB[2][64][40];   // [buf][n][k]
  const int tid  = threadIdx.x;
  const int lane = tid & 31;
  const int wave = tid >> 5;
  const int wM   = wave & 3;
  const int wN   = wave >> 2;
  const int ln   = lane & 15;
  const int half = lane >> 4;
  const int halfK = half * 8;
  const long tileM = (long)blockIdx.y * 128;
  const long tileN = (long)blockIdx.x * 64;

  v8f acc[2][2];
#pragma unroll
  for (int i = 0; i < 2; ++i)
#pragma unroll
    for (int j = 0; j < 2; ++j) acc[i][j] = (v8f){0,0,0,0,0,0,0,0};

  // async tile fetch: A 128x32 = 512 16B-chunks (2/thread), B 64x32 = 256 (1/thread)
  const int ac0 = tid * 2;
  const int br  = tid >> 2;        // 0..63
  const int bcc = tid & 3;         // 0..3

  const int nk = K / 32;
  // prologue: tile 0 -> buf 0
  {
#pragma unroll
    for (int q = 0; q < 2; ++q) {
      int c = ac0 + q, row = c >> 2, cc = c & 3;
      async_ld16(A + (tileM + row) * (long)K + cc * 8, &sA[0][row][cc * 8]);
    }
    async_ld16(Wt + (tileN + br) * (long)K + bcc * 8, &sB[0][br][bcc * 8]);
  }

  for (int t = 0; t < nk; ++t) {
    wait_async0();
    __syncthreads();
    if (t + 1 < nk) {
      const int nb = (t + 1) & 1;
      const int kt = (t + 1) * 32;
#pragma unroll
      for (int q = 0; q < 2; ++q) {
        int c = ac0 + q, row = c >> 2, cc = c & 3;
        async_ld16(A + (tileM + row) * (long)K + kt + cc * 8, &sA[nb][row][cc * 8]);
      }
      async_ld16(Wt + (tileN + br) * (long)K + kt + bcc * 8, &sB[nb][br][bcc * 8]);
    }
    const int buf = t & 1;

    v16bf afr[2];
#pragma unroll
    for (int mi = 0; mi < 2; ++mi) {
      union { v16bf v; u16 u[16]; } f;
      const int row = wM * 32 + mi * 16 + ln;
#pragma unroll
      for (int i = 0; i < 8; ++i) {
        f.u[i]     = sA[buf][row][half * 16 + i];      // A: K {0..7}/{16..23}
        f.u[8 + i] = sA[buf][row][half * 16 + 8 + i];  //    K {8..15}/{24..31}
      }
      afr[mi] = f.v;
    }
#pragma unroll
    for (int ni = 0; ni < 2; ++ni) {
      union { v16bf v; u16 u[16]; } g;
      const int col = wN * 32 + ni * 16 + ln;
#pragma unroll
      for (int i = 0; i < 16; ++i) g.u[i] = sB[buf][col][half * 16 + i];  // B: K run
#pragma unroll
      for (int mi = 0; mi < 2; ++mi)
        acc[mi][ni] = wmma_bf16(afr[mi], g.v, acc[mi][ni]);
    }
    __syncthreads();
  }

#pragma unroll
  for (int mi = 0; mi < 2; ++mi)
#pragma unroll
    for (int ni = 0; ni < 2; ++ni)
#pragma unroll
      for (int r = 0; r < 8; ++r) {
        long row = tileM + wM * 32 + mi * 16 + r + half * 8;
        long col = tileN + wN * 32 + ni * 16 + ln;
        if (OUT_BF16) ((u16*)Cp)[row * (long)N + col] = f2bf(acc[mi][ni][r]);
        else          ((float*)Cp)[row * (long)N + col] = acc[mi][ni][r];
      }
}

// ---------------------------------------------------------------------------
// RoPE (interleaved pairs) + scatter [B,S,NH,HD] bf16 -> [B,NH,S,HD] bf16
// ---------------------------------------------------------------------------
__global__ void rope_scatter(const u16* __restrict__ proj,
                             u16* __restrict__ out, int NH) {
  long idx = (long)blockIdx.x * blockDim.x + threadIdx.x;
  long total = (long)BB * SS * NH * (HDIM / 2);
  if (idx >= total) return;
  int d2 = (int)(idx % (HDIM / 2)); long t = idx / (HDIM / 2);
  int h = (int)(t % NH); t /= NH;
  int s = (int)(t % SS); int b = (int)(t / SS);
  float fr = (float)s * __powf(10000.0f, -2.0f * (float)d2 / (float)HDIM);
  float sn, cs;
  __sincosf(fr, &sn, &cs);
  const u16* p = proj + (((long)(b * SS + s) * NH + h) * HDIM) + 2 * d2;
  float a0 = __uint_as_float((unsigned)p[0] << 16);
  float a1 = __uint_as_float((unsigned)p[1] << 16);
  u16* o = out + ((((long)b * NH + h) * SS + s) * HDIM) + 2 * d2;
  o[0] = f2bf(a0 * cs - a1 * sn);
  o[1] = f2bf(a0 * sn + a1 * cs);
}

// [B,S,KV,HD] bf16 -> [B,KV,HD,S] bf16 (V transposed: P@V B-frags contiguous)
__global__ void v_transpose(const u16* __restrict__ proj, u16* __restrict__ out) {
  long idx = (long)blockIdx.x * blockDim.x + threadIdx.x;
  long total = (long)BB * SS * KVH * HDIM;
  if (idx >= total) return;
  int d  = (int)(idx % HDIM); long t = idx / HDIM;
  int kv = (int)(t % KVH); t /= KVH;
  int s  = (int)(t % SS); int b = (int)(t / SS);
  out[(((long)b * KVH + kv) * HDIM + d) * SS + s] = proj[idx];
}

// ---------------------------------------------------------------------------
// Flash attention: block = (b, h, 128-row q block), 8 waves x 16 q rows.
// Causal mask computed inline (== reference's additive -1e9 mask).
// ---------------------------------------------------------------------------
__global__ __launch_bounds__(256, 1)
void flash_attn(const u16* __restrict__ Qr, const u16* __restrict__ Kr,
                const u16* __restrict__ Vt, u16* __restrict__ O) {
  __shared__ u16 sP[8][16][72];  // per-wave 16x64 bf16 P tile (+pad)
  const int nQB = SS / 128;
  int blk = blockIdx.x;
  int qb = blk % nQB; int t2 = blk / nQB;
  int h = t2 % HH;    int b  = t2 / HH;
  int kvh = h / REP;
  const int lane = threadIdx.x & 31;
  const int wave = threadIdx.x >> 5;
  const int ln   = lane & 15;
  const int half = lane >> 4;
  const int halfK = half * 8;
  const int q0 = qb * 128 + wave * 16;

  const u16* Qbase = Qr + ((long)(b * HH  + h)   * SS) * HDIM;
  const u16* Kbase = Kr + ((long)(b * KVH + kvh) * SS) * HDIM;
  const u16* Vbase = Vt + ((long)(b * KVH + kvh) * HDIM) * SS;

  v16bf qf[4];
#pragma unroll
  for (int dc = 0; dc < 4; ++dc) {
    union { v16bf v; u16 u[16]; } f;
    const u16* p = Qbase + (long)(q0 + ln) * HDIM + dc * 32;
#pragma unroll
    for (int i = 0; i < 8; ++i) { f.u[i] = p[half * 16 + i]; f.u[8 + i] = p[half * 16 + 8 + i]; }
    qf[dc] = f.v;
  }

  v8f oacc[8];
#pragma unroll
  for (int j = 0; j < 8; ++j) oacc[j] = (v8f){0,0,0,0,0,0,0,0};
  float rm[8], rl[8];
#pragma unroll
  for (int r = 0; r < 8; ++r) { rm[r] = -3.0e38f; rl[r] = 0.0f; }

  const float scale = 0.08838834764831845f;  // 1/sqrt(128)
  const int nkv = (qb + 1) * 2;              // 64-wide KV tiles up to block diag
  for (int kt = 0; kt < nkv; ++kt) {
    const int kb = kt * 64;

    // ---- S = Q @ K^T over HD=128 (4 WMMAs per 16x16 score tile) ----
    v8f sacc[4];
#pragma unroll
    for (int ks = 0; ks < 4; ++ks) {
      v8f a = (v8f){0,0,0,0,0,0,0,0};
      const u16* kp = Kbase + (long)(kb + ks * 16 + ln) * HDIM;
#pragma unroll
      for (int dc = 0; dc < 4; ++dc) {
        union { v16bf v; u16 u[16]; } g;
#pragma unroll
        for (int i = 0; i < 16; ++i) g.u[i] = kp[dc * 32 + half * 16 + i];
        a = wmma_bf16(qf[dc], g.v, a);
      }
      sacc[ks] = a;
    }

    // ---- scale + causal mask + online softmax ----
#pragma unroll
    for (int r = 0; r < 8; ++r) {
      int qrow = q0 + r + half * 8;
      float mx = -3.0e38f;
#pragma unroll
      for (int ks = 0; ks < 4; ++ks) {
        int col = kb + ks * 16 + ln;
        float s = sacc[ks][r] * scale + (col > qrow ? -1.0e9f : 0.0f);
        sacc[ks][r] = s;
        mx = fmaxf(mx, s);
      }
      mx = red_max16(mx);
      float rm2   = fmaxf(rm[r], mx);
      float alpha = __expf(rm[r] - rm2);
      rm[r] = rm2;
      rl[r] *= alpha;
#pragma unroll
      for (int j = 0; j < 8; ++j) oacc[j][r] *= alpha;
      float ps = 0.0f;
#pragma unroll
      for (int ks = 0; ks < 4; ++ks) {
        float p = __expf(sacc[ks][r] - rm2);
        sacc[ks][r] = p;
        ps += p;
      }
      rl[r] += red_sum16(ps);
    }

    // ---- re-fragment P via LDS, then O += P @ V ----
    __syncthreads();
#pragma unroll
    for (int ks = 0; ks < 4; ++ks)
#pragma unroll
      for (int r = 0; r < 8; ++r)
        sP[wave][r + half * 8][ks * 16 + ln] = f2bf(sacc[ks][r]);
    __syncthreads();

#pragma unroll
    for (int kc = 0; kc < 2; ++kc) {
      union { v16bf v; u16 u[16]; } f;
#pragma unroll
      for (int i = 0; i < 8; ++i) {
        f.u[i]     = sP[wave][ln][kc * 32 + half * 16 + i];
        f.u[8 + i] = sP[wave][ln][kc * 32 + half * 16 + 8 + i];
      }
#pragma unroll
      for (int j = 0; j < 8; ++j) {
        union { v16bf v; u16 u[16]; } g;
        const u16* vp = Vbase + (long)(j * 16 + ln) * SS + kb + kc * 32;
#pragma unroll
        for (int i = 0; i < 16; ++i) g.u[i] = vp[half * 16 + i];
        oacc[j] = wmma_bf16(f.v, g.v, oacc[j]);
      }
    }
  }

  // ---- normalize + store O bf16 [B, S, H*HD] ----
#pragma unroll
  for (int r = 0; r < 8; ++r) {
    float inv = 1.0f / rl[r];
    long qrow = q0 + r + half * 8;
#pragma unroll
    for (int j = 0; j < 8; ++j) {
      long col = (long)h * HDIM + j * 16 + ln;
      O[((long)b * SS + qrow) * (long)(HH * HDIM) + col] = f2bf(oacc[j][r] * inv);
    }
  }
}

// ---------------------------------------------------------------------------
extern "C" void kernel_launch(void* const* d_in, const int* in_sizes, int n_in,
                              void* d_out, int out_size, void* d_ws, size_t ws_size,
                              hipStream_t stream) {
  (void)in_sizes; (void)n_in; (void)out_size; (void)ws_size;
  const float* x  = (const float*)d_in[0];
  // d_in[1] = additive causal mask: recomputed inline in flash_attn
  const float* wq = (const float*)d_in[2];
  const float* wk = (const float*)d_in[3];
  const float* wv = (const float*)d_in[4];
  const float* wo = (const float*)d_in[5];
  float* out = (float*)d_out;

  char* ws = (char*)d_ws;
  const long M = (long)BB * SS;  // 4096
  u16* xb  = (u16*)ws; ws += M * (long)DDIM * 2;
  u16* wqt = (u16*)ws; ws += (long)(HH  * HDIM) * DDIM * 2;   // [N,K]
  u16* wkt = (u16*)ws; ws += (long)(KVH * HDIM) * DDIM * 2;
  u16* wvt = (u16*)ws; ws += (long)(KVH * HDIM) * DDIM * 2;
  u16* wot = (u16*)ws; ws += (long)DDIM * (HH * HDIM) * 2;
  u16* qpb = (u16*)ws; ws += M * (long)(HH  * HDIM) * 2;      // [B,S,H,HD]
  u16* kpb = (u16*)ws; ws += M * (long)(KVH * HDIM) * 2;
  u16* vpb = (u16*)ws; ws += M * (long)(KVH * HDIM) * 2;
  u16* Qr  = (u16*)ws; ws += (long)BB * HH  * SS * HDIM * 2;  // [B,H,S,HD]
  u16* Kr  = (u16*)ws; ws += (long)BB * KVH * SS * HDIM * 2;  // [B,KV,S,HD]
  u16* Vt  = (u16*)ws; ws += (long)BB * KVH * HDIM * SS * 2;  // [B,KV,HD,S]
  u16* Obf = (u16*)ws;                                        // [B,S,H*HD]

  dim3 blk(256);
  // Precision staging: x -> bf16; weights -> bf16 transposed [N,K]
  long nx = M * (long)DDIM;
  to_bf16<<<dim3((unsigned)((nx + 255) / 256)), blk, 0, stream>>>(x, xb, nx);
  long nwq = (long)DDIM * (HH * HDIM), nwk = (long)DDIM * (KVH * HDIM);
  w_transpose_bf16<<<dim3((unsigned)((nwq + 255) / 256)), blk, 0, stream>>>(wq, wqt, DDIM, HH * HDIM);
  w_transpose_bf16<<<dim3((unsigned)((nwk + 255) / 256)), blk, 0, stream>>>(wk, wkt, DDIM, KVH * HDIM);
  w_transpose_bf16<<<dim3((unsigned)((nwk + 255) / 256)), blk, 0, stream>>>(wv, wvt, DDIM, KVH * HDIM);
  w_transpose_bf16<<<dim3((unsigned)((nwq + 255) / 256)), blk, 0, stream>>>(wo, wot, HH * HDIM, DDIM);

  // Projections (bf16 WMMA GEMMs, async double-buffered tiles)
  gemm_bf16<true><<<dim3((HH  * HDIM) / 64, M / 128), blk, 0, stream>>>(xb, wqt, qpb, (int)M, HH  * HDIM, DDIM);
  gemm_bf16<true><<<dim3((KVH * HDIM) / 64, M / 128), blk, 0, stream>>>(xb, wkt, kpb, (int)M, KVH * HDIM, DDIM);
  gemm_bf16<true><<<dim3((KVH * HDIM) / 64, M / 128), blk, 0, stream>>>(xb, wvt, vpb, (int)M, KVH * HDIM, DDIM);

  // RoPE + attention-friendly layouts
  long nq = (long)BB * SS * HH  * (HDIM / 2);
  long nk = (long)BB * SS * KVH * (HDIM / 2);
  long nv = (long)BB * SS * KVH * HDIM;
  rope_scatter<<<dim3((unsigned)((nq + 255) / 256)), blk, 0, stream>>>(qpb, Qr, HH);
  rope_scatter<<<dim3((unsigned)((nk + 255) / 256)), blk, 0, stream>>>(kpb, Kr, KVH);
  v_transpose <<<dim3((unsigned)((nv + 255) / 256)), blk, 0, stream>>>(vpb, Vt);

  // Flash attention
  flash_attn<<<dim3(BB * HH * (SS / 128)), blk, 0, stream>>>(Qr, Kr, Vt, Obf);

  // Output projection -> fp32
  gemm_bf16<false><<<dim3(DDIM / 64, M / 128), blk, 0, stream>>>(Obf, wot, out, (int)M, DDIM, HH * HDIM);
}
